// MultiQueryAttention_87299505258905
// MI455X (gfx1250) — matrix operand order repaired and checked by reading, not compiled, not run on hardware
//
#include <hip/hip_runtime.h>
#include <hip/hip_bf16.h>

// ---------------------------------------------------------------------------
// MQA forward for MI455X (gfx1250, wave32, WMMA).
// All GEMMs in bf16 (v_wmma_f32_16x16x32_bf16, f32 accumulate) — compute-bound
// (~146 GFLOP vs ~5us of HBM traffic at 23.3 TB/s), so everything routes
// through the matrix pipes. Tile staging now uses the CDNA5 async
// global->LDS path (GLOBAL_LOAD_ASYNC_TO_LDS_B128, ASYNCcnt) instead of a
// VGPR round-trip. Workspace usage ~70.3 MB.
// ---------------------------------------------------------------------------

typedef __bf16 bf16;
typedef __attribute__((ext_vector_type(16))) __bf16 v16bf;
typedef __attribute__((ext_vector_type(8)))  __bf16 v8bf;
typedef __attribute__((ext_vector_type(8)))  float  v8f;

__device__ __forceinline__ v8f vzero8() {
  v8f z = {0.f, 0.f, 0.f, 0.f, 0.f, 0.f, 0.f, 0.f};
  return z;
}

// Async copy 16B global -> LDS (per lane). Generic LDS pointers keep the LDS
// byte offset in addr[31:0] (flat LDS aperture, ISA 10.2), which is what the
// VDST operand of global_load_async_to_lds_* expects.
__device__ __forceinline__ void async_ld_b128(void* lds, const void* g) {
  uint32_t l = (uint32_t)(uintptr_t)lds;
  uint64_t a = (uint64_t)(uintptr_t)g;
  asm volatile("global_load_async_to_lds_b128 %0, %1, off"
               :: "v"(l), "v"(a) : "memory");
}
__device__ __forceinline__ void wait_async() {
  asm volatile("s_wait_asynccnt 0x0" ::: "memory");
}

// Combine two contiguous 8x-bf16 (16B) chunks into one 16-element fragment.
__device__ __forceinline__ v16bf frag_pair(const bf16* p0, const bf16* p1) {
  union { v16bf v; v8bf h[2]; } u;
  u.h[0] = *(const v8bf*)p0;
  u.h[1] = *(const v8bf*)p1;
  return u.v;
}

// A-matrix fragment (16x32, M x K), row-major source with leading dim ld.
// ISA layout: lanes 0-15 hold K {0..7,16..23}; lanes 16-31 hold K {8..15,24..31}.
__device__ __forceinline__ v16bf afrag(const bf16* base, int ld, int row, int kb, int hi) {
  const bf16* p = base + (size_t)row * ld + kb + hi * 8;
  return frag_pair(p, p + 16);
}

// B-matrix fragment (32x16, K x N), N-major source (row = N, contiguous K).
// ISA layout: lanes 0-15 hold K {0..7,16..23}; lanes 16-31 hold K {8..15,24..31} for col N=lane%16.
__device__ __forceinline__ v16bf bfrag(const bf16* base, int ld, int col, int kb, int hi) {
  const bf16* p = base + (size_t)col * ld + kb + hi * 8;
  return frag_pair(p, p + 16);
}

__device__ __forceinline__ v8f wmma_bf16f32(v16bf a, v16bf b, v8f c) {
  // D = A*B + C, 16x16x32 bf16 -> f32
  return __builtin_amdgcn_wmma_f32_16x16x32_bf16(false, a, false, b, (short)0, c,
                                                 false, false);
}

// ---------------------------------------------------------------------------
// Elementwise f32 -> bf16 convert.
__global__ void cvt_bf16(const float* __restrict__ in, bf16* __restrict__ out, int n) {
  int i = blockIdx.x * blockDim.x + threadIdx.x;
  if (i < n) out[i] = (bf16)in[i];
}

// Transpose-convert: in[R][C] f32 -> out[C][R] bf16 (weights become N-major).
__global__ void tcvt_bf16(const float* __restrict__ in, bf16* __restrict__ out,
                          int R, int C) {
  int c = blockIdx.x * blockDim.x + threadIdx.x;
  int r = blockIdx.y * blockDim.y + threadIdx.y;
  if (r < R && c < C) out[(size_t)c * R + r] = (bf16)in[(size_t)r * C + c];
}

// ---------------------------------------------------------------------------
// WMMA GEMM: C = A[M,K] * Bt[N,K]^T + bias, block tile 128x128, K-tile 64.
// 8 waves: wave (w&3) -> 32-row strip, (w>>2) -> 64-col strip; 2x4 wmma tiles/wave.
// MODE 0: bf16 out (scaled). MODE 1: f32 out. MODE 2: bf16 transposed out [N][ld_t].
template <int MODE>
__global__ __launch_bounds__(256) void gemm_wmma(
    const bf16* __restrict__ A, const bf16* __restrict__ Bt,
    const float* __restrict__ bias, void* __restrict__ out,
    int M, int N, int K, float scale, int ld_t) {
  constexpr int BK = 64, LDT = 72;  // LDS row padded: 144B per row (16B multiple)
  __shared__ alignas(16) bf16 Alds[128 * LDT];
  __shared__ alignas(16) bf16 Blds[128 * LDT];

  const int tid = threadIdx.x, lane = tid & 31, w = tid >> 5;
  const int lm = lane & 15, hi = lane >> 4;
  const int wm = (w & 3) * 32, wn = (w >> 2) * 64;
  const int m0 = blockIdx.y * 128, n0 = blockIdx.x * 128;

  v8f acc[2][4];
#pragma unroll
  for (int i = 0; i < 2; ++i)
#pragma unroll
    for (int j = 0; j < 4; ++j) acc[i][j] = vzero8();

  // Cooperative staging: 256 threads x 32 bf16 per tile (A and B each),
  // moved with async global->LDS b128 transfers.
  const int lrow = tid >> 1;
  const int lcol = (tid & 1) * 32;
  const bf16* gA = A + (size_t)(m0 + lrow) * K + lcol;
  const bf16* gB = Bt + (size_t)(n0 + lrow) * K + lcol;
  bf16* lA = Alds + lrow * LDT + lcol;
  bf16* lB = Blds + lrow * LDT + lcol;

  for (int kt = 0; kt < K; kt += BK) {
#pragma unroll
    for (int j = 0; j < 4; ++j) {
      async_ld_b128(lA + j * 8, gA + kt + j * 8);
      async_ld_b128(lB + j * 8, gB + kt + j * 8);
    }
    wait_async();
    __syncthreads();
#pragma unroll
    for (int kc = 0; kc < 2; ++kc) {
      const int kb = kc * 32;
      v16bf af[2], bfv[4];
#pragma unroll
      for (int mt = 0; mt < 2; ++mt) af[mt] = afrag(Alds, LDT, wm + mt * 16 + lm, kb, hi);
#pragma unroll
      for (int nt = 0; nt < 4; ++nt) bfv[nt] = bfrag(Blds, LDT, wn + nt * 16 + lm, kb, hi);
#pragma unroll
      for (int mt = 0; mt < 2; ++mt)
#pragma unroll
        for (int nt = 0; nt < 4; ++nt)
          acc[mt][nt] = wmma_bf16f32(af[mt], bfv[nt], acc[mt][nt]);
    }
    __syncthreads();
  }

#pragma unroll
  for (int mt = 0; mt < 2; ++mt)
#pragma unroll
    for (int nt = 0; nt < 4; ++nt)
#pragma unroll
      for (int r = 0; r < 8; ++r) {
        int row = m0 + wm + mt * 16 + r + hi * 8;
        int col = n0 + wn + nt * 16 + lm;
        float v = acc[mt][nt][r] + bias[col];
        if (MODE == 0)
          ((bf16*)out)[(size_t)row * N + col] = (bf16)(v * scale);
        else if (MODE == 1)
          ((float*)out)[(size_t)row * N + col] = v;
        else
          ((bf16*)out)[(size_t)col * ld_t + row] = (bf16)(v * scale);
      }
}

// ---------------------------------------------------------------------------
// Flash MQA: grid (S/64 q-tiles, H, B), 128 threads (4 waves x 16 q-rows).
// Q pre-scaled by 1/sqrt(dk). K stored [B*S][128]; V stored transposed [128][B*S].
__global__ __launch_bounds__(128) void mqa_flash(
    const bf16* __restrict__ Q, const bf16* __restrict__ Kg,
    const bf16* __restrict__ Vt, const int* __restrict__ mask,
    bf16* __restrict__ attn) {
  constexpr int S = 2048, DK = 128, D = 2048;
  constexpr int LDK = 136, LDV = 72, LDP = 72;  // padded, rows stay 16B multiples
  __shared__ alignas(16) bf16 Klds[64 * LDK];    // 64 keys x 128 d
  __shared__ alignas(16) bf16 Vlds[128 * LDV];   // 128 d x 64 keys (transposed)
  __shared__ alignas(16) bf16 Plds[4 * 16 * LDP];// per-wave P staging (C->A layout)

  const int b = blockIdx.z, h = blockIdx.y;
  const int q0 = blockIdx.x * 64;
  const int tid = threadIdx.x, lane = tid & 31, w = tid >> 5;
  const int lm = lane & 15, hi = lane >> 4;
  const int qm = q0 + w * 16;

  // Q fragments for this wave's 16 rows: 4 K-chunks of 32 over d_k=128.
  const bf16* qrowp = Q + (size_t)(b * S + qm + lm) * D + h * DK;
  v16bf qf[4];
#pragma unroll
  for (int kc = 0; kc < 4; ++kc) {
    const bf16* p = qrowp + kc * 32 + hi * 8;
    qf[kc] = frag_pair(p, p + 16);
  }

  v8f acc[8];
  float m_run[8], l_run[8];
#pragma unroll
  for (int o = 0; o < 8; ++o) acc[o] = vzero8();
#pragma unroll
  for (int r = 0; r < 8; ++r) { m_run[r] = -__builtin_inff(); l_run[r] = 0.f; }

  bf16* Pw = Plds + w * 16 * LDP;
  const int krow = tid >> 1, kcol = (tid & 1) * 64;
  const bf16* gK = Kg + ((size_t)b * S + krow) * DK + kcol;
  const bf16* gV = Vt + (size_t)tid * ((size_t)2 * S) + (size_t)b * S;
  const int* mrow = mask + (size_t)b * S * S;

  for (int k0 = 0; k0 < S; k0 += 64) {
    // Stage K tile (64x128) and V^T tile (128x64) into LDS via async copies.
#pragma unroll
    for (int j = 0; j < 8; ++j)
      async_ld_b128(Klds + krow * LDK + kcol + j * 8, gK + (size_t)k0 * DK + j * 8);
#pragma unroll
    for (int j = 0; j < 8; ++j)
      async_ld_b128(Vlds + tid * LDV + j * 8, gV + k0 + j * 8);
    wait_async();
    __syncthreads();

    // Scores: 16 q-rows x 64 keys = 4 N-tiles x 4 K-chunks of WMMA.
    v8f sc[4];
#pragma unroll
    for (int nt = 0; nt < 4; ++nt) {
      sc[nt] = vzero8();
#pragma unroll
      for (int kc = 0; kc < 4; ++kc)
        sc[nt] = wmma_bf16f32(qf[kc], bfrag(Klds, LDK, nt * 16 + lm, kc * 32, hi), sc[nt]);
    }

    // Mask (element (q,k) lives at lane N=lm, reg r with M = r + 8*hi).
#pragma unroll
    for (int nt = 0; nt < 4; ++nt)
#pragma unroll
      for (int r = 0; r < 8; ++r) {
        int qg = qm + r + hi * 8;
        int kg = k0 + nt * 16 + lm;
        if (mrow[(size_t)qg * S + kg] == 0) sc[nt][r] = -__builtin_inff();
      }

    // Online softmax: row reductions across the 16 lanes of each half.
    float mnew[8], alpha[8], rs[8];
#pragma unroll
    for (int r = 0; r < 8; ++r) {
      float t = fmaxf(fmaxf(sc[0][r], sc[1][r]), fmaxf(sc[2][r], sc[3][r]));
      t = fmaxf(t, __shfl_xor(t, 1));
      t = fmaxf(t, __shfl_xor(t, 2));
      t = fmaxf(t, __shfl_xor(t, 4));
      t = fmaxf(t, __shfl_xor(t, 8));
      float mn = fmaxf(m_run[r], t);
      alpha[r] = __expf(m_run[r] - mn);
      mnew[r] = mn;
      m_run[r] = mn;
      rs[r] = 0.f;
    }
#pragma unroll
    for (int nt = 0; nt < 4; ++nt)
#pragma unroll
      for (int r = 0; r < 8; ++r) {
        float pv = __expf(sc[nt][r] - mnew[r]);
        sc[nt][r] = pv;
        rs[r] += pv;
      }
#pragma unroll
    for (int r = 0; r < 8; ++r) {
      float s = rs[r];
      s += __shfl_xor(s, 1);
      s += __shfl_xor(s, 2);
      s += __shfl_xor(s, 4);
      s += __shfl_xor(s, 8);
      l_run[r] = l_run[r] * alpha[r] + s;
    }
#pragma unroll
    for (int o = 0; o < 8; ++o)
#pragma unroll
      for (int r = 0; r < 8; ++r) acc[o][r] *= alpha[r];

    // C-layout -> A-layout via wave-private LDS staging.
#pragma unroll
    for (int nt = 0; nt < 4; ++nt)
#pragma unroll
      for (int r = 0; r < 8; ++r)
        Pw[(r + hi * 8) * LDP + nt * 16 + lm] = (bf16)sc[nt][r];
    asm volatile("s_wait_dscnt 0x0" ::: "memory");

    // PV: 8 d-tiles x 2 K-chunks of 32 keys.
#pragma unroll
    for (int kc = 0; kc < 2; ++kc) {
      v16bf pf = afrag(Pw, LDP, lm, kc * 32, hi);
#pragma unroll
      for (int o = 0; o < 8; ++o)
        acc[o] = wmma_bf16f32(pf, bfrag(Vlds, LDV, o * 16 + lm, kc * 32, hi), acc[o]);
    }
    __syncthreads();
  }

  float inv[8];
#pragma unroll
  for (int r = 0; r < 8; ++r) inv[r] = 1.f / l_run[r];
#pragma unroll
  for (int o = 0; o < 8; ++o)
#pragma unroll
    for (int r = 0; r < 8; ++r) {
      int qg = qm + r + hi * 8;
      attn[(size_t)(b * S + qg) * D + h * DK + o * 16 + lm] = (bf16)(acc[o][r] * inv[r]);
    }
}

// ---------------------------------------------------------------------------
extern "C" void kernel_launch(void* const* d_in, const int* in_sizes, int n_in,
                              void* d_out, int out_size, void* d_ws, size_t ws_size,
                              hipStream_t stream) {
  (void)in_sizes; (void)n_in; (void)out_size; (void)ws_size;
  const float* x   = (const float*)d_in[0];
  const int*   msk = (const int*)d_in[1];
  const float* w_q = (const float*)d_in[2];
  const float* b_q = (const float*)d_in[3];
  const float* w_k = (const float*)d_in[4];
  const float* b_k = (const float*)d_in[5];
  const float* w_v = (const float*)d_in[6];
  const float* b_v = (const float*)d_in[7];
  const float* w_o = (const float*)d_in[8];
  const float* b_o = (const float*)d_in[9];
  float* out = (float*)d_out;

  constexpr int B = 2, S = 2048, D = 2048, H = 16, DK = 128;
  constexpr size_t MS = (size_t)B * S;  // 4096 rows

  char* p = (char*)d_ws;
  bf16* x_bf = (bf16*)p;  p += MS * D * 2;          // 16.8 MB
  bf16* wqT  = (bf16*)p;  p += (size_t)D * D * 2;   //  8.4 MB
  bf16* wkT  = (bf16*)p;  p += (size_t)D * DK * 2;  //  0.5 MB
  bf16* wvT  = (bf16*)p;  p += (size_t)D * DK * 2;  //  0.5 MB
  bf16* woT  = (bf16*)p;  p += (size_t)D * D * 2;   //  8.4 MB
  bf16* Qb   = (bf16*)p;  p += MS * D * 2;          // 16.8 MB (pre-scaled)
  bf16* Kb   = (bf16*)p;  p += MS * DK * 2;         //  1.0 MB
  bf16* Vtb  = (bf16*)p;  p += MS * DK * 2;         //  1.0 MB ([dk][B*S])
  bf16* Ab   = (bf16*)p;  p += MS * D * 2;          // 16.8 MB

  const int nx = (int)(MS * D);
  cvt_bf16<<<(nx + 255) / 256, 256, 0, stream>>>(x, x_bf, nx);
  tcvt_bf16<<<dim3(D / 32, D / 8), dim3(32, 8), 0, stream>>>(w_q, wqT, D, D);
  tcvt_bf16<<<dim3(DK / 32, D / 8), dim3(32, 8), 0, stream>>>(w_k, wkT, D, DK);
  tcvt_bf16<<<dim3(DK / 32, D / 8), dim3(32, 8), 0, stream>>>(w_v, wvT, D, DK);
  tcvt_bf16<<<dim3(D / 32, D / 8), dim3(32, 8), 0, stream>>>(w_o, woT, D, D);

  const float qscale = 0.08838834764831845f;  // 1/sqrt(128), folded into Q
  gemm_wmma<0><<<dim3(D / 128, MS / 128), 256, 0, stream>>>(
      x_bf, wqT, b_q, Qb, (int)MS, D, D, qscale, 0);
  gemm_wmma<0><<<dim3(1, MS / 128), 256, 0, stream>>>(
      x_bf, wkT, b_k, Kb, (int)MS, DK, D, 1.f, 0);
  gemm_wmma<2><<<dim3(1, MS / 128), 256, 0, stream>>>(
      x_bf, wvT, b_v, Vtb, (int)MS, DK, D, 1.f, (int)MS);
  mqa_flash<<<dim3(S / 64, H, B), 128, 0, stream>>>(Qb, Kb, Vtb, msk, Ab);
  gemm_wmma<1><<<dim3(D / 128, MS / 128), 256, 0, stream>>>(
      Ab, woT, b_o, out, (int)MS, D, D, 1.f, 0);
}
